// DecoderLayer_31988916420901
// MI455X (gfx1250) — matrix operand order repaired and checked by reading, not compiled, not run on hardware
//
#include <hip/hip_runtime.h>
#include <hip/hip_bf16.h>
#include <math.h>

// ---------------------------------------------------------------------------
// Types for CDNA5 WMMA (wave32): 16x16x32 bf16 -> f32
// ---------------------------------------------------------------------------
typedef __bf16 bf16;
typedef __attribute__((ext_vector_type(16))) __bf16 bf16x16;
typedef __attribute__((ext_vector_type(8)))  __bf16 bf16x8;
typedef __attribute__((ext_vector_type(8)))  float  f32x8;

__device__ __forceinline__ f32x8 wmma_bf16f32(bf16x16 a, bf16x16 b, f32x8 c) {
  // 8 args: (neg_a, A, neg_b, B, c_mod, C, reuse_a, reuse_b)
  return __builtin_amdgcn_wmma_f32_16x16x32_bf16(false, a, false, b, (short)0, c,
                                                 false, false);
}

// A fragment (16x32 bf16, M x K): lane l holds row m=l&15, half h=l>>4.
// elems 0..7  = K[k0 + h*8 .. +7], elems 8..15 = K[k0+16+h*8 .. +7].
// caller passes p = rowptr + k0 + h*8  (contiguous 16B + 16B loads)
__device__ __forceinline__ bf16x16 frag_a_bf16(const bf16* p) {
  bf16x8 lo = *reinterpret_cast<const bf16x8*>(p);
  bf16x8 hi = *reinterpret_cast<const bf16x8*>(p + 16);
  bf16x16 r;
#pragma unroll
  for (int i = 0; i < 8; ++i) { r[i] = lo[i]; r[8 + i] = hi[i]; }
  return r;
}

// Same A layout but source data is fp32 (attention probabilities) -> cvt.
__device__ __forceinline__ bf16x16 frag_a_f32(const float* p) {
  bf16x16 r;
#pragma unroll
  for (int i = 0; i < 8; ++i) { r[i] = (bf16)p[i]; r[8 + i] = (bf16)p[16 + i]; }
  return r;
}

// B fragment (32x16 bf16, K x N): lane l holds column n=l&15, half h=l>>4,
// elems 0..7 = K[k0 + h*8 .. +7], elems 8..15 = K[k0+16+h*8 .. +7]. Source layout is B^T row-major (one row
// per output column, contiguous over K) -> 16B + 16B loads per lane.
// caller passes p = colptr + k0 + h*8
__device__ __forceinline__ bf16x16 frag_b_bf16(const bf16* p) {
  bf16x8 lo = *reinterpret_cast<const bf16x8*>(p);
  bf16x8 hi = *reinterpret_cast<const bf16x8*>(p + 16);
  bf16x16 r;
#pragma unroll
  for (int i = 0; i < 8; ++i) { r[i] = lo[i]; r[8 + i] = hi[i]; }
  return r;
}

// ---------------------------------------------------------------------------
// Generic GEMM:  C[M,N] = A[M,K] (bf16, row-major) @ Bt[N,K]^T (bf16) + bias
// 32x32 output tile per wave = 2x2 WMMA sub-tiles (register blocking: 4 WMMAs
// per 4 fragment loads -> 2x the arithmetic intensity of a 16x16 tile).
// 8 waves per block. Optional ReLU; outputs fp32 and/or bf16; optional "VT"
// bf16 store producing per-head-transposed V layout [(b*H+h)*DK+dk]*T + t.
// ---------------------------------------------------------------------------
template <bool RELU, bool STORE_VT>
__global__ __launch_bounds__(256) void gemm_bf16_kernel(
    const bf16* __restrict__ A, const bf16* __restrict__ Bt,
    const float* __restrict__ bias,
    float* __restrict__ Cf, bf16* __restrict__ Cb,
    int M, int N, int K, int lda, int ldb, int ldc,
    int Tdim, int Hdim, int DKdim) {
  const int lane = threadIdx.x & 31;
  const int hl   = lane >> 4;
  const int wv   = threadIdx.x >> 5;
  const int tnc  = N >> 5;
  const long tile = (long)blockIdx.x * (blockDim.x >> 5) + wv;
  if (tile >= (long)(M >> 5) * tnc) return;
  const int m0 = (int)(tile / tnc) << 5;
  const int n0 = (int)(tile % tnc) << 5;

  const bf16* arow[2] = {A  + (size_t)(m0 +      (lane & 15)) * lda,
                         A  + (size_t)(m0 + 16 + (lane & 15)) * lda};
  const bf16* brow[2] = {Bt + (size_t)(n0 +      (lane & 15)) * ldb,
                         Bt + (size_t)(n0 + 16 + (lane & 15)) * ldb};

  f32x8 acc[2][2] = {};
  for (int k0 = 0; k0 < K; k0 += 32) {
    if (k0 + 32 < K) {  // emits global_prefetch_b8
      __builtin_prefetch(arow[0] + k0 + 32, 0, 1);
      __builtin_prefetch(arow[1] + k0 + 32, 0, 1);
      __builtin_prefetch(brow[0] + k0 + 32, 0, 1);
      __builtin_prefetch(brow[1] + k0 + 32, 0, 1);
    }
    bf16x16 af[2] = {frag_a_bf16(arow[0] + k0 + hl * 8),
                     frag_a_bf16(arow[1] + k0 + hl * 8)};
    bf16x16 bf[2] = {frag_b_bf16(brow[0] + k0 + hl * 8),
                     frag_b_bf16(brow[1] + k0 + hl * 8)};
#pragma unroll
    for (int i = 0; i < 2; ++i)
#pragma unroll
      for (int j = 0; j < 2; ++j)
        acc[i][j] = wmma_bf16f32(af[i], bf[j], acc[i][j]);
  }

#pragma unroll
  for (int i = 0; i < 2; ++i) {
#pragma unroll
    for (int j = 0; j < 2; ++j) {
      const int col = n0 + j * 16 + (lane & 15);
      const float bv = bias ? bias[col] : 0.f;
#pragma unroll
      for (int r = 0; r < 8; ++r) {
        const int row = m0 + i * 16 + r + 8 * hl;  // C/D layout: halves M, M+8
        float v = acc[i][j][r] + bv;
        if (RELU) v = fmaxf(v, 0.f);
        if (Cf) Cf[(size_t)row * ldc + col] = v;
        if (Cb) {
          if (STORE_VT) {
            const int b = row / Tdim, t = row % Tdim;
            const int h = col / DKdim, dk = col % DKdim;
            Cb[(((size_t)b * Hdim + h) * DKdim + dk) * Tdim + t] = (bf16)v;
          } else {
            Cb[(size_t)row * ldc + col] = (bf16)v;
          }
        }
      }
    }
  }
}

// ---------------------------------------------------------------------------
// scores[b,h,q,k] = (Q[b,q,h,:] . K[b,k,h,:]) * scale, masked; 32x32 tile
// per wave (2x2 WMMA). mask_mode 0: mask[q*Tk+k] (causal); 1: mask[b*Tk+k].
// K rows are contiguous over the head dim, so they ARE the B^T layout.
// ---------------------------------------------------------------------------
__global__ __launch_bounds__(256) void attn_scores_kernel(
    const bf16* __restrict__ Q, const bf16* __restrict__ Kmat,
    const int* __restrict__ mask, int mask_mode,
    float* __restrict__ out,
    int Tq, int Tk, int Ddim, int Hdim, int DKdim, float scale) {
  const int lane = threadIdx.x & 31, hl = lane >> 4, wv = threadIdx.x >> 5;
  const int bh = blockIdx.z;
  const int b = bh / Hdim, h = bh % Hdim;
  const int tkc = Tk >> 5;
  const long tile = (long)blockIdx.x * (blockDim.x >> 5) + wv;
  if (tile >= (long)(Tq >> 5) * tkc) return;
  const int q0  = (int)(tile / tkc) << 5;
  const int k0c = (int)(tile % tkc) << 5;

  const bf16* qrow[2] = {
      Q + ((size_t)b * Tq + q0 +      (lane & 15)) * Ddim + (size_t)h * DKdim,
      Q + ((size_t)b * Tq + q0 + 16 + (lane & 15)) * Ddim + (size_t)h * DKdim};
  const bf16* krow[2] = {
      Kmat + ((size_t)b * Tk + k0c +      (lane & 15)) * Ddim + (size_t)h * DKdim,
      Kmat + ((size_t)b * Tk + k0c + 16 + (lane & 15)) * Ddim + (size_t)h * DKdim};

  f32x8 acc[2][2] = {};
  for (int kk = 0; kk < DKdim; kk += 32) {
    bf16x16 af[2] = {frag_a_bf16(qrow[0] + kk + hl * 8),
                     frag_a_bf16(qrow[1] + kk + hl * 8)};
    bf16x16 bf[2] = {frag_b_bf16(krow[0] + kk + hl * 8),
                     frag_b_bf16(krow[1] + kk + hl * 8)};
#pragma unroll
    for (int i = 0; i < 2; ++i)
#pragma unroll
      for (int j = 0; j < 2; ++j)
        acc[i][j] = wmma_bf16f32(af[i], bf[j], acc[i][j]);
  }

  float* obase = out + ((size_t)b * Hdim + h) * Tq * (size_t)Tk;
#pragma unroll
  for (int i = 0; i < 2; ++i) {
#pragma unroll
    for (int j = 0; j < 2; ++j) {
      const int col = k0c + j * 16 + (lane & 15);
#pragma unroll
      for (int r = 0; r < 8; ++r) {
        const int row = q0 + i * 16 + r + 8 * hl;
        float v = acc[i][j][r] * scale;
        const int mv = (mask_mode == 0) ? mask[(size_t)row * Tk + col]
                                        : mask[(size_t)b * Tk + col];
        if (mv == 0) v = -1e9f;
        obase[(size_t)row * Tk + col] = v;
      }
    }
  }
}

// ---------------------------------------------------------------------------
// out[b,q,h*DK+dk] = sum_k P[b,h,q,k] * V[b,k,h,dk].  P fp32 -> bf16 A frags
// on the fly; V stored per-head transposed so B frags are contiguous.
// 32(q) x 32(dk) tile per wave (2x2 WMMA). Output bf16.
// ---------------------------------------------------------------------------
__global__ __launch_bounds__(256) void attn_v_kernel(
    const float* __restrict__ P, const bf16* __restrict__ Vt,
    bf16* __restrict__ Out, int Tq, int L, int Ddim, int Hdim, int DKdim) {
  const int lane = threadIdx.x & 31, hl = lane >> 4, wv = threadIdx.x >> 5;
  const int bh = blockIdx.z;
  const int b = bh / Hdim, h = bh % Hdim;
  const int tnc = DKdim >> 5;
  const long tile = (long)blockIdx.x * (blockDim.x >> 5) + wv;
  if (tile >= (long)(Tq >> 5) * tnc) return;
  const int q0 = (int)(tile / tnc) << 5;
  const int n0 = (int)(tile % tnc) << 5;

  const float* prow[2] = {
      P + (((size_t)b * Hdim + h) * Tq + q0 +      (lane & 15)) * L,
      P + (((size_t)b * Hdim + h) * Tq + q0 + 16 + (lane & 15)) * L};
  const bf16* vrow[2] = {
      Vt + (((size_t)b * Hdim + h) * DKdim + n0 +      (lane & 15)) * L,
      Vt + (((size_t)b * Hdim + h) * DKdim + n0 + 16 + (lane & 15)) * L};

  f32x8 acc[2][2] = {};
  for (int k0 = 0; k0 < L; k0 += 32) {
    if (k0 + 32 < L) {
      __builtin_prefetch(prow[0] + k0 + 32, 0, 1);
      __builtin_prefetch(prow[1] + k0 + 32, 0, 1);
      __builtin_prefetch(vrow[0] + k0 + 32, 0, 1);
      __builtin_prefetch(vrow[1] + k0 + 32, 0, 1);
    }
    bf16x16 af[2] = {frag_a_f32(prow[0] + k0 + hl * 8),
                     frag_a_f32(prow[1] + k0 + hl * 8)};
    bf16x16 bf[2] = {frag_b_bf16(vrow[0] + k0 + hl * 8),
                     frag_b_bf16(vrow[1] + k0 + hl * 8)};
#pragma unroll
    for (int i = 0; i < 2; ++i)
#pragma unroll
      for (int j = 0; j < 2; ++j)
        acc[i][j] = wmma_bf16f32(af[i], bf[j], acc[i][j]);
  }

#pragma unroll
  for (int i = 0; i < 2; ++i) {
#pragma unroll
    for (int j = 0; j < 2; ++j) {
      const int col = n0 + j * 16 + (lane & 15);
#pragma unroll
      for (int r = 0; r < 8; ++r) {
        const int row = q0 + i * 16 + r + 8 * hl;
        Out[((size_t)b * Tq + row) * Ddim + (size_t)h * DKdim + col] =
            (bf16)acc[i][j][r];
      }
    }
  }
}

// ---------------------------------------------------------------------------
// In-place row softmax over length L (these rows are the returned attention
// weights living in d_out). One 256-thread block per row.
// ---------------------------------------------------------------------------
__global__ __launch_bounds__(256) void softmax_kernel(float* __restrict__ data, int L) {
  float* p = data + (size_t)blockIdx.x * L;
  __shared__ float red[256];
  const int tid = threadIdx.x;
  float m = -3.0e38f;
  for (int i = tid; i < L; i += 256) m = fmaxf(m, p[i]);
  red[tid] = m; __syncthreads();
  for (int s = 128; s > 0; s >>= 1) { if (tid < s) red[tid] = fmaxf(red[tid], red[tid + s]); __syncthreads(); }
  m = red[0]; __syncthreads();
  float sum = 0.f;
  for (int i = tid; i < L; i += 256) { float e = __expf(p[i] - m); p[i] = e; sum += e; }
  red[tid] = sum; __syncthreads();
  for (int s = 128; s > 0; s >>= 1) { if (tid < s) red[tid] += red[tid + s]; __syncthreads(); }
  const float inv = 1.f / red[0];
  for (int i = tid; i < L; i += 256) p[i] *= inv;
}

// ---------------------------------------------------------------------------
// y = gamma * (x + r - mean) / (std + eps) + beta, std unbiased (ddof=1).
// Writes fp32 (optional) and bf16 copy (optional) for the next GEMM.
// One 256-thread block per row; D must be 1024.
// ---------------------------------------------------------------------------
__global__ __launch_bounds__(256) void add_ln_kernel(
    const float* __restrict__ x, const float* __restrict__ r,
    const float* __restrict__ g, const float* __restrict__ be,
    float* __restrict__ yf, bf16* __restrict__ yb, int Dd) {
  const size_t row = blockIdx.x;
  const float* xp = x + row * Dd;
  const float* rp = r + row * Dd;
  __shared__ float red[256], red2[256];
  const int tid = threadIdx.x;
  float s = 0.f, ss = 0.f, vloc[4];
#pragma unroll
  for (int i = 0; i < 4; ++i) {
    const int idx = tid + (i << 8);
    const float v = xp[idx] + rp[idx];
    vloc[i] = v; s += v; ss += v * v;
  }
  red[tid] = s; red2[tid] = ss; __syncthreads();
  for (int st = 128; st > 0; st >>= 1) {
    if (tid < st) { red[tid] += red[tid + st]; red2[tid] += red2[tid + st]; }
    __syncthreads();
  }
  const float mean = red[0] / Dd;
  const float var  = (red2[0] - Dd * mean * mean) / (Dd - 1);
  const float inv  = 1.f / (sqrtf(fmaxf(var, 0.f)) + 1e-6f);
#pragma unroll
  for (int i = 0; i < 4; ++i) {
    const int idx = tid + (i << 8);
    const float y = g[idx] * (vloc[i] - mean) * inv + be[idx];
    if (yf) yf[row * Dd + idx] = y;
    if (yb) yb[row * Dd + idx] = (bf16)y;
  }
}

// ---------------------------------------------------------------------------
// fp32 -> bf16 conversions (plain, and transposing for weight matrices so
// GEMM B-operand loads are contiguous: dst[n*K+k] = src[k*N+n]).
// ---------------------------------------------------------------------------
__global__ void cvt_kernel(const float* __restrict__ src, bf16* __restrict__ dst, long n) {
  const long i = (long)blockIdx.x * 256 + threadIdx.x;
  if (i < n) dst[i] = (bf16)src[i];
}
__global__ void cvt_t_kernel(const float* __restrict__ src, bf16* __restrict__ dst,
                             int K, int N) {
  const long i = (long)blockIdx.x * 256 + threadIdx.x;
  if (i >= (long)K * N) return;
  const int n = (int)(i / K), k = (int)(i % K);
  dst[i] = (bf16)src[(size_t)k * N + n];
}

// ---------------------------------------------------------------------------
extern "C" void kernel_launch(void* const* d_in, const int* in_sizes, int n_in,
                              void* d_out, int out_size, void* d_ws, size_t ws_size,
                              hipStream_t stream) {
  (void)in_sizes; (void)n_in; (void)out_size; (void)ws_size;
  const int Bsz = 4, T = 1024, S = 1024, D = 1024, Hh = 16, DFF = 4096, DK = 64;
  const long MT = (long)Bsz * T;  // 4096 rows (also B*S)

  // ---- inputs (setup_inputs dict order) ----
  const float* x    = (const float*)d_in[0];
  const float* enc  = (const float*)d_in[1];
  const int*   srcm = (const int*)d_in[2];
  const int*   tgtm = (const int*)d_in[3];
  const float* wq_s = (const float*)d_in[4];  const float* bq_s = (const float*)d_in[5];
  const float* wk_s = (const float*)d_in[6];  const float* bk_s = (const float*)d_in[7];
  const float* wv_s = (const float*)d_in[8];  const float* bv_s = (const float*)d_in[9];
  const float* wo_s = (const float*)d_in[10]; const float* bo_s = (const float*)d_in[11];
  const float* wq_c = (const float*)d_in[12]; const float* bq_c = (const float*)d_in[13];
  const float* wk_c = (const float*)d_in[14]; const float* bk_c = (const float*)d_in[15];
  const float* wv_c = (const float*)d_in[16]; const float* bv_c = (const float*)d_in[17];
  const float* wo_c = (const float*)d_in[18]; const float* bo_c = (const float*)d_in[19];
  const float* w1   = (const float*)d_in[20]; const float* b1   = (const float*)d_in[21];
  const float* w2   = (const float*)d_in[22]; const float* b2   = (const float*)d_in[23];
  const float* g1   = (const float*)d_in[24]; const float* be1  = (const float*)d_in[25];
  const float* g2   = (const float*)d_in[26]; const float* be2  = (const float*)d_in[27];
  const float* g3   = (const float*)d_in[28]; const float* be3  = (const float*)d_in[29];

  // ---- output regions: (x, self_w, cross_w) concatenated ----
  float* out_x   = (float*)d_out;
  float* self_w  = out_x + (size_t)MT * D;
  float* cross_w = self_w + (size_t)Bsz * Hh * T * T;

  // ---- workspace carve ----
  char* wp = (char*)d_ws;
  auto carve = [&](size_t bytes) -> char* {
    char* p = wp; wp += (bytes + 255) & ~(size_t)255; return p;
  };
  bf16* xb    = (bf16*)carve((size_t)MT * D * 2);
  bf16* encb  = (bf16*)carve((size_t)MT * D * 2);
  bf16* wqst  = (bf16*)carve((size_t)D * D * 2);
  bf16* wkst  = (bf16*)carve((size_t)D * D * 2);
  bf16* wvst  = (bf16*)carve((size_t)D * D * 2);
  bf16* wost  = (bf16*)carve((size_t)D * D * 2);
  bf16* wqct  = (bf16*)carve((size_t)D * D * 2);
  bf16* wkct  = (bf16*)carve((size_t)D * D * 2);
  bf16* wvct  = (bf16*)carve((size_t)D * D * 2);
  bf16* woct  = (bf16*)carve((size_t)D * D * 2);
  bf16* w1t   = (bf16*)carve((size_t)D * DFF * 2);
  bf16* w2t   = (bf16*)carve((size_t)D * DFF * 2);
  bf16* Qs    = (bf16*)carve((size_t)MT * D * 2);
  bf16* Ks    = (bf16*)carve((size_t)MT * D * 2);
  bf16* Vt    = (bf16*)carve((size_t)MT * D * 2);   // per-head transposed V
  bf16* avb   = (bf16*)carve((size_t)MT * D * 2);   // attn@V output, bf16
  float* sa   = (float*)carve((size_t)MT * D * 4);  // attn block fp32 out
  float* x1   = (float*)carve((size_t)MT * D * 4);
  bf16*  x1b  = (bf16*)carve((size_t)MT * D * 2);
  float* x2   = (float*)carve((size_t)MT * D * 4);
  bf16*  x2b  = (bf16*)carve((size_t)MT * D * 2);
  bf16*  hb   = (bf16*)carve((size_t)MT * DFF * 2); // FFN hidden, bf16
  float* ff   = (float*)carve((size_t)MT * D * 4);

  const long nXD = MT * (long)D;
  const unsigned cvtb = (unsigned)((nXD + 255) / 256);
  const unsigned cvtw = (unsigned)(((long)D * D + 255) / 256);
  const unsigned cvtf = (unsigned)(((long)D * DFF + 255) / 256);
  // 32x32 tiles, 8 waves per block
  auto gblk = [](long M, long N) { return (unsigned)(((M >> 5) * (N >> 5)) >> 3); };

  // ---- 1. fp32 -> bf16 conversions (weights transposed) ----
  cvt_kernel<<<cvtb, 256, 0, stream>>>(x,   xb,   nXD);
  cvt_kernel<<<cvtb, 256, 0, stream>>>(enc, encb, nXD);
  cvt_t_kernel<<<cvtw, 256, 0, stream>>>(wq_s, wqst, D, D);
  cvt_t_kernel<<<cvtw, 256, 0, stream>>>(wk_s, wkst, D, D);
  cvt_t_kernel<<<cvtw, 256, 0, stream>>>(wv_s, wvst, D, D);
  cvt_t_kernel<<<cvtw, 256, 0, stream>>>(wo_s, wost, D, D);
  cvt_t_kernel<<<cvtw, 256, 0, stream>>>(wq_c, wqct, D, D);
  cvt_t_kernel<<<cvtw, 256, 0, stream>>>(wk_c, wkct, D, D);
  cvt_t_kernel<<<cvtw, 256, 0, stream>>>(wv_c, wvct, D, D);
  cvt_t_kernel<<<cvtw, 256, 0, stream>>>(wo_c, woct, D, D);
  cvt_t_kernel<<<cvtf, 256, 0, stream>>>(w1, w1t, D, DFF);
  cvt_t_kernel<<<cvtf, 256, 0, stream>>>(w2, w2t, DFF, D);

  // ---- 2. self-attention ----
  gemm_bf16_kernel<false, false><<<gblk(MT, D), 256, 0, stream>>>(
      xb, wqst, bq_s, nullptr, Qs, MT, D, D, D, D, D, 0, 0, 0);
  gemm_bf16_kernel<false, false><<<gblk(MT, D), 256, 0, stream>>>(
      xb, wkst, bk_s, nullptr, Ks, MT, D, D, D, D, D, 0, 0, 0);
  gemm_bf16_kernel<false, true><<<gblk(MT, D), 256, 0, stream>>>(
      xb, wvst, bv_s, nullptr, Vt, MT, D, D, D, D, D, T, Hh, DK);

  // (T/32)*(T/32) = 1024 wave-tiles per head -> 128 blocks of 8 waves
  attn_scores_kernel<<<dim3(128, 1, Bsz * Hh), 256, 0, stream>>>(
      Qs, Ks, tgtm, 0, self_w, T, T, D, Hh, DK, 0.125f);
  softmax_kernel<<<Bsz * Hh * T, 256, 0, stream>>>(self_w, T);
  // (T/32)*(DK/32) = 64 wave-tiles per head -> 8 blocks
  attn_v_kernel<<<dim3(8, 1, Bsz * Hh), 256, 0, stream>>>(
      self_w, Vt, avb, T, T, D, Hh, DK);
  gemm_bf16_kernel<false, false><<<gblk(MT, D), 256, 0, stream>>>(
      avb, wost, bo_s, sa, nullptr, MT, D, D, D, D, D, 0, 0, 0);
  add_ln_kernel<<<(unsigned)MT, 256, 0, stream>>>(x, sa, g1, be1, x1, x1b, D);

  // ---- 3. cross-attention ----
  gemm_bf16_kernel<false, false><<<gblk(MT, D), 256, 0, stream>>>(
      x1b, wqct, bq_c, nullptr, Qs, MT, D, D, D, D, D, 0, 0, 0);
  gemm_bf16_kernel<false, false><<<gblk(MT, D), 256, 0, stream>>>(
      encb, wkct, bk_c, nullptr, Ks, MT, D, D, D, D, D, 0, 0, 0);
  gemm_bf16_kernel<false, true><<<gblk(MT, D), 256, 0, stream>>>(
      encb, wvct, bv_c, nullptr, Vt, MT, D, D, D, D, D, S, Hh, DK);

  attn_scores_kernel<<<dim3(128, 1, Bsz * Hh), 256, 0, stream>>>(
      Qs, Ks, srcm, 1, cross_w, T, S, D, Hh, DK, 0.125f);
  softmax_kernel<<<Bsz * Hh * T, 256, 0, stream>>>(cross_w, S);
  attn_v_kernel<<<dim3(8, 1, Bsz * Hh), 256, 0, stream>>>(
      cross_w, Vt, avb, T, S, D, Hh, DK);
  gemm_bf16_kernel<false, false><<<gblk(MT, D), 256, 0, stream>>>(
      avb, woct, bo_c, sa, nullptr, MT, D, D, D, D, D, 0, 0, 0);
  add_ln_kernel<<<(unsigned)MT, 256, 0, stream>>>(x1, sa, g2, be2, x2, x2b, D);

  // ---- 4. feed-forward ----
  gemm_bf16_kernel<true, false><<<gblk(MT, DFF), 256, 0, stream>>>(
      x2b, w1t, b1, nullptr, hb, MT, DFF, D, D, D, DFF, 0, 0, 0);
  gemm_bf16_kernel<false, false><<<gblk(MT, D), 256, 0, stream>>>(
      hb, w2t, b2, ff, nullptr, MT, D, DFF, DFF, DFF, D, 0, 0, 0);
  add_ln_kernel<<<(unsigned)MT, 256, 0, stream>>>(x2, ff, g3, be3, out_x, nullptr, D);
}
